// AttentionBlock_41154376630422
// MI455X (gfx1250) — matrix operand order repaired and checked by reading, not compiled, not run on hardware
//
#include <hip/hip_runtime.h>
#include <hip/hip_bf16.h>

// Problem constants (B=2, S=2048, D=1024, H=16, HD=64)
#define S_LEN   2048
#define NHEAD   16
#define DMODEL  1024
#define HDIM    64
#define NTOK    4096            // B * S
#define NPROJ   3072            // H * 3 * HD

typedef __attribute__((ext_vector_type(16))) _Float16 v16h;
typedef __attribute__((ext_vector_type(8)))  _Float16 v8h;
typedef __attribute__((ext_vector_type(8)))  float    v8f;

static __device__ inline v8f wmma_f16(v16h a, v16h b, v8f c) {
  // D = A(16x32 f16) * B(32x16 f16) + C(16x16 f32)
  return __builtin_amdgcn_wmma_f32_16x16x32_f16(false, a, false, b,
                                                (short)0, c, false, false);
}

// inf-safe fast tanh: 1 - 2/(e^2x + 1)  (saturates to +/-1)
static __device__ inline float fast_tanh(float x) {
  const float t = __expf(2.f * x);
  return 1.f - 2.f * __builtin_amdgcn_rcpf(t + 1.f);
}

// ---- fragment builders (wave32 layouts per CDNA5 ISA 7.12.2) -------------
// A 16x32 f16: lanes 0-15 row M=lane, halves 0..7 = K0..7, 8..15 = K16..23;
//              lanes 16-31 same rows, K8..15 / K24..31.
static __device__ inline v16h frag_a_lds(const _Float16* t, int ld) {
  const int lane = threadIdx.x & 31;
  const int row  = lane & 15;
  const int k0   = (lane >> 4) << 3;          // 0 or 8
  const _Float16* p = t + row * ld + k0;
  v8h lo = *(const v8h*)(p);
  v8h hi = *(const v8h*)(p + 16);
  v16h a;
#pragma unroll
  for (int i = 0; i < 8; ++i) { a[i] = lo[i]; a[8 + i] = hi[i]; }
  return a;
}

// B 32x16 f16 from a row-major NxK tile (operand stored transposed in LDS):
// lanes 0-15 hold K0..7 / K16..23, lanes 16-31 hold K8..15 / K24..31; N = lane&15.
// Two 8-half runs per lane -> two ds_load_b128.
static __device__ inline v16h frag_b_nxk(const _Float16* t, int ld) {
  const int lane = threadIdx.x & 31;
  const int n  = lane & 15;
  const int k0 = (lane >> 4) << 3;            // 0 or 8
  const _Float16* p = t + n * ld + k0;
  v8h lo = *(const v8h*)(p);
  v8h hi = *(const v8h*)(p + 16);
  v16h b;
#pragma unroll
  for (int i = 0; i < 8; ++i) { b[i] = lo[i]; b[8 + i] = hi[i]; }
  return b;
}

// ==========================================================================
// Kernel 1: fused QKV projection + RoPE.
// GEMM [4096 x 1024] * [1024 x 3072], 64(M) x 64(N) per block, 8 waves as
// 4(M) x 2(N); each wave owns two 16x16 C tiles (2 WMMA per 32-K chunk).
// Each 64-wide N tile is exactly q, k, or v of one head (192 = 3*64), so the
// RoPE pair (d, d^32) lives inside this block's tile.
// ==========================================================================
__global__ __launch_bounds__(256)
void qkv_rope_kernel(const float* __restrict__ x, const int* __restrict__ pos,
                     const float* __restrict__ w_in,
                     _Float16* __restrict__ Qh, _Float16* __restrict__ Kh,
                     _Float16* __restrict__ Vh) {
  __shared__ __align__(16) _Float16 lds_a [64 * 40];  // 64 tokens x 32 K
  __shared__ __align__(16) _Float16 lds_bt[64 * 40];  // transposed: 64 N x 32 K
  __shared__ __align__(16) float    lds_p [64 * 68];  // raw proj tile

  const int tid  = threadIdx.x;
  const int wave = tid >> 5;
  const int wm   = wave & 3;                   // 0..3 (16-row tiles)
  const int wn   = wave >> 2;                  // 0..1 (32-col halves)
  const int m0   = blockIdx.x * 64;            // token base
  const int n0   = blockIdx.y * 64;            // proj-column base

  v8f acc0 = {}, acc1 = {};
  for (int kc = 0; kc < DMODEL / 32; ++kc) {
    __syncthreads();
    {   // stage A: 64x32 fp32 -> f16 (vector LDS store)
      const int r = tid >> 2, c8 = (tid & 3) * 8;
      const float* src = x + (size_t)(m0 + r) * DMODEL + kc * 32 + c8;
      const float4 a0 = *(const float4*)(src);
      const float4 a1 = *(const float4*)(src + 4);
      v8h t;
      t[0] = (_Float16)a0.x; t[1] = (_Float16)a0.y;
      t[2] = (_Float16)a0.z; t[3] = (_Float16)a0.w;
      t[4] = (_Float16)a1.x; t[5] = (_Float16)a1.y;
      t[6] = (_Float16)a1.z; t[7] = (_Float16)a1.w;
      *(v8h*)(lds_a + r * 40 + c8) = t;
    }
    {   // stage B transposed: w_in[32 K x 64 N] -> lds_bt[N][K]
      const int r = tid >> 3, c8 = (tid & 7) * 8;
      const float* src = w_in + (size_t)(kc * 32 + r) * NPROJ + n0 + c8;
      const float4 b0 = *(const float4*)(src);
      const float4 b1 = *(const float4*)(src + 4);
      lds_bt[(c8 + 0) * 40 + r] = (_Float16)b0.x;
      lds_bt[(c8 + 1) * 40 + r] = (_Float16)b0.y;
      lds_bt[(c8 + 2) * 40 + r] = (_Float16)b0.z;
      lds_bt[(c8 + 3) * 40 + r] = (_Float16)b0.w;
      lds_bt[(c8 + 4) * 40 + r] = (_Float16)b1.x;
      lds_bt[(c8 + 5) * 40 + r] = (_Float16)b1.y;
      lds_bt[(c8 + 6) * 40 + r] = (_Float16)b1.z;
      lds_bt[(c8 + 7) * 40 + r] = (_Float16)b1.w;
    }
    __syncthreads();
    v16h a = frag_a_lds(lds_a + wm * 16 * 40, 40);
    acc0 = wmma_f16(a, frag_b_nxk(lds_bt + (wn * 32 +  0) * 40, 40), acc0);
    acc1 = wmma_f16(a, frag_b_nxk(lds_bt + (wn * 32 + 16) * 40, 40), acc1);
  }
  __syncthreads();
  {   // C tiles -> LDS (row = 8*(lane>=16)+g, col = lane&15)
    const int lane = tid & 31;
    const int n = lane & 15, mb = (lane >> 4) * 8;
#pragma unroll
    for (int g = 0; g < 8; ++g) {
      lds_p[(wm * 16 + mb + g) * 68 + wn * 32 + n]      = acc0[g];
      lds_p[(wm * 16 + mb + g) * 68 + wn * 32 + 16 + n] = acc1[g];
    }
  }
  __syncthreads();

  // Epilogue: RoPE (+ 1/sqrt(64) on q), scatter to [B,H,S,64] f16.
  const int h   = n0 / 192;
  const int typ = (n0 % 192) / 64;             // 0=q 1=k 2=v
  const int r   = tid >> 2;
  const int c16 = (tid & 3) * 16;
  const int tok = m0 + r;
  const int bi  = tok / S_LEN, s = tok % S_LEN;
  const int p   = pos[bi * S_LEN + s];
  _Float16* dst = (typ == 0) ? Qh : (typ == 1) ? Kh : Vh;
  const size_t base = (((size_t)bi * NHEAD + h) * S_LEN + s) * HDIM;
#pragma unroll
  for (int i = 0; i < 16; ++i) {
    const int d = c16 + i;
    const float v = lds_p[r * 68 + d];
    float o;
    if (typ == 2) {
      o = v;
    } else {
      const int j = d & 31;                    // rotation pair index
      // 1/timescale = 10000^(-j/32) = exp(-j * ln(1e4)/32)
      const float inv_ts = __expf((float)j * -0.287823137f);
      const float ang = (float)p * inv_ts;
      float sn, cs;
      __sincosf(ang, &sn, &cs);
      const float other = lds_p[r * 68 + (d ^ 32)];
      o = (d < 32) ? (v * cs - other * sn) : (v * cs + other * sn);
      if (typ == 0) o *= 0.125f;               // 1/sqrt(HD)
    }
    dst[base + d] = (_Float16)o;
  }
}

// ==========================================================================
// Kernel 2: flash attention, one block per (b, h, 64-query tile).
// 8 waves = 4(M-rows of 16) x 2(N-halves of 32). Online softmax in LDS.
// Masked logits are exactly -50 (mask-before-softcap semantics); the
// all-masked tail beyond the diagonal contributes n_tail*e^(-50-m) to the
// denominator (numerator tail ~1e-20 relative: dropped).
// ==========================================================================
__global__ __launch_bounds__(256)
void flash_attn_kernel(const _Float16* __restrict__ Qh,
                       const _Float16* __restrict__ Kh,
                       const _Float16* __restrict__ Vh,
                       _Float16* __restrict__ Xh) {
  __shared__ __align__(16) _Float16 lds_q [64 * 72];  // [query][hd]
  __shared__ __align__(16) _Float16 lds_k [64 * 72];  // [key][hd]
  __shared__ __align__(16) _Float16 lds_vt[64 * 72];  // transposed: [hd][key]
  __shared__ __align__(16) _Float16 lds_pm[64 * 72];  // [query][key] probs f16
  __shared__ __align__(16) float    lds_s [64 * 68];  // logits f32
  __shared__ float row_m[64], row_l[64], row_al[64], part[64 * 4];

  const int tid  = threadIdx.x;
  const int wave = tid >> 5;
  const int wm   = wave & 3;                   // 0..3
  const int wn   = wave >> 2;                  // 0..1
  const int qb = blockIdx.x, h = blockIdx.y, bi = blockIdx.z;
  const int q0 = qb * 64;
  const size_t head = (((size_t)bi * NHEAD + h) * S_LEN) * HDIM;

  {   // stage Q tile (64x64 f16), 32B per thread
    const int r = tid >> 2, c = (tid & 3) * 16;
    const v8h* src = (const v8h*)(Qh + head + (size_t)(q0 + r) * HDIM + c);
    v8h* d = (v8h*)(lds_q + r * 72 + c);
    d[0] = src[0]; d[1] = src[1];
  }
  if (tid < 64) { row_m[tid] = -1e30f; row_l[tid] = 0.f; }
  __syncthreads();

  v16h qf[2];
  qf[0] = frag_a_lds(lds_q + wm * 16 * 72 + 0, 72);
  qf[1] = frag_a_lds(lds_q + wm * 16 * 72 + 32, 72);

  v8f o0 = {}, o1 = {};                        // O accumulators (2 hd-subtiles)
  const int nblocks = qb + 1;

  for (int kb = 0; kb < nblocks; ++kb) {
    __syncthreads();                           // prior-iteration LDS reads done
    {   // stage K (row-major) and V (transposed) blocks
      const int r = tid >> 2, c = (tid & 3) * 16;
      const size_t g = head + (size_t)(kb * 64 + r) * HDIM + c;
      const v8h* sk = (const v8h*)(Kh + g);
      const v8h* sv = (const v8h*)(Vh + g);
      v8h* dk = (v8h*)(lds_k + r * 72 + c);
      dk[0] = sk[0]; dk[1] = sk[1];
      const v8h v0 = sv[0], v1 = sv[1];
#pragma unroll
      for (int i = 0; i < 8; ++i) {
        lds_vt[(c + i)     * 72 + r] = v0[i];
        lds_vt[(c + 8 + i) * 72 + r] = v1[i];
      }
      if (kb + 1 < nblocks) {                  // global_prefetch_b8 next block
        __builtin_prefetch(Kh + g + 64 * HDIM, 0, 0);
        __builtin_prefetch(Vh + g + 64 * HDIM, 0, 0);
      }
    }
    __syncthreads();

    // ---- S = Q K^T, soft-cap + causal mask, spill to lds_s ----
#pragma unroll
    for (int nsub = 0; nsub < 2; ++nsub) {
      const int nt = wn * 32 + nsub * 16;
      v8f sa = {};
      sa = wmma_f16(qf[0], frag_b_nxk(lds_k + nt * 72 + 0, 72), sa);
      sa = wmma_f16(qf[1], frag_b_nxk(lds_k + nt * 72 + 32, 72), sa);
      const int lane = tid & 31;
      const int n = lane & 15, mb = (lane >> 4) * 8;
#pragma unroll
      for (int g = 0; g < 8; ++g) {
        const int row = wm * 16 + mb + g;
        const int key = kb * 64 + nt + n;
        float sv = 50.f * fast_tanh(sa[g] * 0.02f);
        if (key > q0 + row) sv = -50.f;        // mask-then-cap == exactly -50
        lds_s[row * 68 + nt + n] = sv;
      }
    }
    __syncthreads();

    // ---- online softmax: partial max ----
    {
      const int r = tid >> 2, q4 = tid & 3;
      float pm = -1e30f;
#pragma unroll
      for (int i = 0; i < 16; ++i) pm = fmaxf(pm, lds_s[r * 68 + q4 * 16 + i]);
      part[r * 4 + q4] = pm;
    }
    __syncthreads();
    if (tid < 64) {
      const float m_old = row_m[tid];
      float m_new = fmaxf(fmaxf(part[tid * 4 + 0], part[tid * 4 + 1]),
                          fmaxf(part[tid * 4 + 2], part[tid * 4 + 3]));
      m_new = fmaxf(m_old, m_new);
      row_al[tid] = __expf(m_old - m_new);
      row_m[tid] = m_new;
    }
    __syncthreads();
    {   // P = exp(s - m) -> f16 tile + partial sums
      const int r = tid >> 2, q4 = tid & 3;
      const float mr = row_m[r];
      float ps = 0.f;
#pragma unroll
      for (int i = 0; i < 16; ++i) {
        const float pe = __expf(lds_s[r * 68 + q4 * 16 + i] - mr);
        lds_pm[r * 72 + q4 * 16 + i] = (_Float16)pe;
        ps += pe;
      }
      part[r * 4 + q4] = ps;
    }
    __syncthreads();
    if (tid < 64)
      row_l[tid] = row_l[tid] * row_al[tid] + part[tid * 4 + 0] +
                   part[tid * 4 + 1] + part[tid * 4 + 2] + part[tid * 4 + 3];

    // ---- rescale O by alpha, then O += P V ----
    {
      const int lane = tid & 31;
      const int mb = (lane >> 4) * 8;
#pragma unroll
      for (int g = 0; g < 8; ++g) {
        const float al = row_al[wm * 16 + mb + g];
        o0[g] *= al; o1[g] *= al;
      }
    }
    v16h pa0 = frag_a_lds(lds_pm + wm * 16 * 72 + 0, 72);
    v16h pa1 = frag_a_lds(lds_pm + wm * 16 * 72 + 32, 72);
    {
      const int nt0 = wn * 32, nt1 = wn * 32 + 16;
      o0 = wmma_f16(pa0, frag_b_nxk(lds_vt + nt0 * 72 + 0, 72), o0);
      o0 = wmma_f16(pa1, frag_b_nxk(lds_vt + nt0 * 72 + 32, 72), o0);
      o1 = wmma_f16(pa0, frag_b_nxk(lds_vt + nt1 * 72 + 0, 72), o1);
      o1 = wmma_f16(pa1, frag_b_nxk(lds_vt + nt1 * 72 + 32, 72), o1);
    }
  }
  __syncthreads();

  // ---- normalize (with masked-tail denominator) and emit X[B,S,H,64] f16 --
  {
    const int lane = tid & 31;
    const int n = lane & 15, mb = (lane >> 4) * 8;
    const float tail = (float)(S_LEN - nblocks * 64);
#pragma unroll
    for (int g = 0; g < 8; ++g) {
      const int row = wm * 16 + mb + g;
      const float den = row_l[row] + tail * __expf(-50.f - row_m[row]);
      const float inv = __builtin_amdgcn_rcpf(den);
      const size_t rb = ((size_t)(bi * S_LEN + q0 + row)) * DMODEL + h * HDIM;
      Xh[rb + wn * 32 + n]      = (_Float16)(o0[g] * inv);
      Xh[rb + wn * 32 + 16 + n] = (_Float16)(o1[g] * inv);
    }
  }
}

// ==========================================================================
// Kernel 3: output projection [4096 x 1024] * [1024 x 1024] -> fp32 out.
// 64x64 block tiles, 8 waves as 4(M) x 2(N), 2 WMMA per chunk per wave.
// ==========================================================================
__global__ __launch_bounds__(256)
void out_proj_kernel(const _Float16* __restrict__ Xh,
                     const float* __restrict__ w_out,
                     float* __restrict__ out) {
  __shared__ __align__(16) _Float16 lds_a [64 * 40];
  __shared__ __align__(16) _Float16 lds_bt[64 * 40];  // transposed [N][K]

  const int tid  = threadIdx.x;
  const int wave = tid >> 5;
  const int wm = wave & 3, wn = wave >> 2;
  const int m0 = blockIdx.x * 64, n0 = blockIdx.y * 64;

  v8f acc0 = {}, acc1 = {};
  for (int kc = 0; kc < DMODEL / 32; ++kc) {
    __syncthreads();
    {   // stage A (already f16): one 16B vector copy per thread
      const int r = tid >> 2, c8 = (tid & 3) * 8;
      *(v8h*)(lds_a + r * 40 + c8) =
          *(const v8h*)(Xh + (size_t)(m0 + r) * DMODEL + kc * 32 + c8);
    }
    {   // stage B transposed: w_out fp32 -> f16
      const int r = tid >> 3, c8 = (tid & 7) * 8;
      const float* src = w_out + (size_t)(kc * 32 + r) * DMODEL + n0 + c8;
      const float4 b0 = *(const float4*)(src);
      const float4 b1 = *(const float4*)(src + 4);
      lds_bt[(c8 + 0) * 40 + r] = (_Float16)b0.x;
      lds_bt[(c8 + 1) * 40 + r] = (_Float16)b0.y;
      lds_bt[(c8 + 2) * 40 + r] = (_Float16)b0.z;
      lds_bt[(c8 + 3) * 40 + r] = (_Float16)b0.w;
      lds_bt[(c8 + 4) * 40 + r] = (_Float16)b1.x;
      lds_bt[(c8 + 5) * 40 + r] = (_Float16)b1.y;
      lds_bt[(c8 + 6) * 40 + r] = (_Float16)b1.z;
      lds_bt[(c8 + 7) * 40 + r] = (_Float16)b1.w;
    }
    __syncthreads();
    v16h a = frag_a_lds(lds_a + wm * 16 * 40, 40);
    acc0 = wmma_f16(a, frag_b_nxk(lds_bt + (wn * 32 +  0) * 40, 40), acc0);
    acc1 = wmma_f16(a, frag_b_nxk(lds_bt + (wn * 32 + 16) * 40, 40), acc1);
  }
  {
    const int lane = tid & 31;
    const int n = lane & 15, mb = (lane >> 4) * 8;
#pragma unroll
    for (int g = 0; g < 8; ++g) {
      const size_t rb = (size_t)(m0 + wm * 16 + mb + g) * DMODEL + n0 + wn * 32;
      out[rb + n]      = acc0[g];
      out[rb + 16 + n] = acc1[g];
    }
  }
}

// ==========================================================================
extern "C" void kernel_launch(void* const* d_in, const int* in_sizes, int n_in,
                              void* d_out, int out_size, void* d_ws, size_t ws_size,
                              hipStream_t stream) {
  (void)in_sizes; (void)n_in; (void)out_size; (void)ws_size;
  const float* x     = (const float*)d_in[0];       // [B,S,D] fp32
  const int*   pos   = (const int*)d_in[1];         // [B,S] int32
  // d_in[2] = causal mask (tril by construction; realized via key<=query)
  const float* w_in  = (const float*)d_in[3];       // [D,H,3*HD] fp32
  const float* w_out = (const float*)d_in[4];       // [H,HD,D] fp32
  float* out = (float*)d_out;                       // [B,S,D] fp32

  const size_t nelem = (size_t)2 * NHEAD * S_LEN * HDIM;   // 4,194,304
  _Float16* Qh = (_Float16*)d_ws;
  _Float16* Kh = Qh + nelem;
  _Float16* Vh = Kh + nelem;
  _Float16* Xh = Vh + nelem;                        // [B*S, D] f16

  qkv_rope_kernel<<<dim3(NTOK / 64, NPROJ / 64), 256, 0, stream>>>(
      x, pos, w_in, Qh, Kh, Vh);
  flash_attn_kernel<<<dim3(S_LEN / 64, NHEAD, 2), 256, 0, stream>>>(
      Qh, Kh, Vh, Xh);
  out_proj_kernel<<<dim3(NTOK / 64, DMODEL / 64), 256, 0, stream>>>(
      Xh, w_out, out);
}
